// BiLSTM_CRF_72232759984195
// MI455X (gfx1250) — matrix operand order repaired and checked by reading, not compiled, not run on hardware
//
#include <hip/hip_runtime.h>
#include <hip/hip_bf16.h>

typedef __attribute__((ext_vector_type(16))) __bf16 v16bf;
typedef __attribute__((ext_vector_type(8)))  __bf16 bf16x8;
typedef __attribute__((ext_vector_type(8)))  float  v8f;

// ---------------- workspace layout (bytes, all 256B aligned) ----------------
static constexpr size_t SZ_EMB   = (size_t)512 * 64 * 256 * 2;   // 16 MiB (bf16, [S][B][E])
static constexpr size_t SZ_W     = (size_t)1024 * 256 * 2;       // 512 KiB (bf16, row-major)
static constexpr size_t SZ_W2T   = (size_t)32 * 512 * 2;         // 32 KiB
static constexpr size_t SZ_LSTM  = (size_t)64 * 512 * 512 * 2;   // 32 MiB (bf16, [B][S][512])
static constexpr size_t SZ_FEATS = (size_t)64 * 512 * 32 * 4;    // 4 MiB (f32)
static constexpr size_t SZ_BP    = (size_t)64 * 512 * 32;        // 1 MiB (u8)

static constexpr size_t OFF_EMBF  = 0;
static constexpr size_t OFF_EMBB  = OFF_EMBF + SZ_EMB;
static constexpr size_t OFF_WIHF  = OFF_EMBB + SZ_EMB;
static constexpr size_t OFF_WHHF  = OFF_WIHF + SZ_W;
static constexpr size_t OFF_WIHB  = OFF_WHHF + SZ_W;
static constexpr size_t OFF_WHHB  = OFF_WIHB + SZ_W;
static constexpr size_t OFF_W2T   = OFF_WHHB + SZ_W;
static constexpr size_t OFF_LSTM  = OFF_W2T + SZ_W2T;
static constexpr size_t OFF_FEATS = OFF_LSTM + SZ_LSTM;
static constexpr size_t OFF_BP    = OFF_FEATS + SZ_FEATS;

__device__ __forceinline__ float sigf(float x) { return 1.0f / (1.0f + __expf(-x)); }

__device__ __forceinline__ v16bf load_a_frag(const __bf16* ap) {
    bf16x8 lo = *(const bf16x8*)ap;
    bf16x8 hi = *(const bf16x8*)(ap + 8);
    v16bf a;
#pragma unroll
    for (int e = 0; e < 8; ++e) { a[e] = lo[e]; a[e + 8] = hi[e]; }
    return a;
}

// ---------------- fp32 -> bf16 conversion ----------------
__global__ void cvt_kernel(const float* __restrict__ in, __bf16* __restrict__ out, int n) {
    int i = blockIdx.x * blockDim.x + threadIdx.x;
    if (i < n) out[i] = (__bf16)in[i];
}

// ---------------- embedding gather (+ reversed copy for backward LSTM) ----------------
// embF/embB layout: [s][b][e], bf16
__global__ void embed_kernel(const int* __restrict__ sent, const int* __restrict__ lengths,
                             const float* __restrict__ table,
                             __bf16* __restrict__ embF, __bf16* __restrict__ embB) {
    int gidx = blockIdx.x * blockDim.x + threadIdx.x;   // S*B*32 threads, 8 elems each
    int row  = gidx >> 5;                               // row = s*64 + b
    int eo   = (gidx & 31) * 8;
    int s = row >> 6;
    int b = row & 63;
    int len  = lengths[b];
    int tokF = sent[b * 512 + s];
    int srev = (s < len) ? (len - 1 - s) : s;
    int tokB = sent[b * 512 + srev];
    const float* pf = table + (size_t)tokF * 256 + eo;
    const float* pb = table + (size_t)tokB * 256 + eo;
    bf16x8 vf, vb;
#pragma unroll
    for (int i = 0; i < 8; ++i) { vf[i] = (__bf16)pf[i]; vb[i] = (__bf16)pb[i]; }
    size_t o = (size_t)row * 256 + eo;
    *(bf16x8*)(embF + o) = vf;
    *(bf16x8*)(embB + o) = vb;
}

// ---------------- persistent recurrent LSTM (one WGP per direction) ----------------
// grid=2 (dir), block=1024 (32 waves). Per step: gates(64x1024) = [x_t | h] @ [Wih Whh]^T
// Batch processed in two halves of 32 to bound LDS (h 32KB + gates_T 64KB + len).
__global__ __launch_bounds__(1024) void lstm_kernel(
    const __bf16* __restrict__ embF, const __bf16* __restrict__ embB,
    const __bf16* __restrict__ wihF, const __bf16* __restrict__ whhF,
    const __bf16* __restrict__ wihB, const __bf16* __restrict__ whhB,
    const float* __restrict__ bihF, const float* __restrict__ bhhF,
    const float* __restrict__ bihB, const float* __restrict__ bhhB,
    const float* __restrict__ h0, const float* __restrict__ c0,
    const int* __restrict__ lengths, __bf16* __restrict__ lstm_out) {
    __shared__ alignas(16) __bf16 h_lds[64 * 256];      // 32 KB: h state, bf16, row-major [b][j]
    __shared__ alignas(16) __bf16 gates_T[1024 * 32];   // 64 KB: gate staging, TRANSPOSED [n][m]
    __shared__ int len_lds[64];

    const int dir = blockIdx.x;
    const __bf16* emb = dir ? embB : embF;
    const __bf16* wih = dir ? wihB : wihF;
    const __bf16* whh = dir ? whhB : whhF;
    const float* bih = dir ? bihB : bihF;
    const float* bhh = dir ? bhhB : bhhF;

    const int tid   = threadIdx.x;
    const int lane  = tid & 31;
    const int wave  = tid >> 5;
    const int mtl   = wave >> 4;            // 0..1 : local m-tile within half
    const int ntB   = (wave & 15) * 4;      // 4 n-tiles per wave
    const int kHalf = lane >> 4;            // 0/1
    const int n0    = lane & 15;
    const int mLane = lane & 15;

    // elementwise mapping: thread owns column jj for batches b = half*32 + k*4 + bq
    const int jj = tid & 255;
    const int bq = tid >> 8;                // 0..3
    float c_reg[16];
    float bias_g[4];
#pragma unroll
    for (int g = 0; g < 4; ++g) bias_g[g] = bih[g * 256 + jj] + bhh[g * 256 + jj];
#pragma unroll
    for (int k = 0; k < 16; ++k) {
        int b = (k >> 3) * 32 + (k & 7) * 4 + bq;   // k = half*8 + kk
        c_reg[k] = c0[dir * 16384 + b * 256 + jj];
    }
    for (int k = 0; k < 16; ++k) {
        int idx = tid + k * 1024;
        h_lds[idx] = (__bf16)h0[dir * 16384 + idx];
    }
    if (tid < 64) len_lds[tid] = lengths[tid];
    __syncthreads();

#pragma unroll 1
    for (int t = 0; t < 512; ++t) {
        const __bf16* xbase = emb + (size_t)t * 64 * 256;
#pragma unroll 1
        for (int half = 0; half < 2; ++half) {
            // -------- GEMM: 32x1024 tile block, K = 256(x) + 256(h) --------
            const int mRow = half * 32 + mtl * 16 + mLane;   // global batch row
            v8f acc[4] = {};
            // x-phase: A from global emb, B from Wih  (pure global loads)
            const __bf16* xrow = xbase + (size_t)mRow * 256 + kHalf * 16;
#pragma unroll 1
            for (int kk = 0; kk < 8; ++kk) {
                const int kb32 = kk * 32;
                v16bf a = load_a_frag(xrow + kb32);
                const int kb = kb32 + kHalf * 16;
#pragma unroll
                for (int j = 0; j < 4; ++j) {
                    const int nrow = (ntB + j) * 16 + n0;    // gate column
                    v16bf bfrag = *(const v16bf*)(wih + (size_t)nrow * 256 + kb);
                    acc[j] = __builtin_amdgcn_wmma_f32_16x16x32_bf16(
                        false, a, false, bfrag, (short)0, acc[j], false, false);
                }
            }
            // h-phase: A from LDS h state, B from Whh  (pure ds loads for A)
            const __bf16* hrow = &h_lds[mRow * 256 + kHalf * 16];
#pragma unroll 1
            for (int kk = 0; kk < 8; ++kk) {
                const int kb32 = kk * 32;
                v16bf a = load_a_frag(hrow + kb32);
                const int kb = kb32 + kHalf * 16;
#pragma unroll
                for (int j = 0; j < 4; ++j) {
                    const int nrow = (ntB + j) * 16 + n0;
                    v16bf bfrag = *(const v16bf*)(whh + (size_t)nrow * 256 + kb);
                    acc[j] = __builtin_amdgcn_wmma_f32_16x16x32_bf16(
                        false, a, false, bfrag, (short)0, acc[j], false, false);
                }
            }
            // store D tiles to LDS transposed [n][m]: each lane's 8 rows are contiguous
            const int mBase = mtl * 16 + kHalf * 8;          // local batch row base (0..31)
#pragma unroll
            for (int j = 0; j < 4; ++j) {
                const int ncol = (ntB + j) * 16 + n0;
                bf16x8 pk;
#pragma unroll
                for (int r = 0; r < 8; ++r) pk[r] = (__bf16)acc[j][r];
                *(bf16x8*)&gates_T[ncol * 32 + mBase] = pk;  // one ds_store_b128
            }
            __syncthreads();
            // -------- elementwise LSTM cell update for this half --------
#pragma unroll 1
            for (int k = 0; k < 8; ++k) {
                const int bl = k * 4 + bq;               // local batch 0..31
                const int b  = half * 32 + bl;
                const int len = len_lds[b];
                const bool m = (t < len);
                float gi = (float)gates_T[jj * 32 + bl]          + bias_g[0];
                float gf = (float)gates_T[(256 + jj) * 32 + bl]  + bias_g[1];
                float gg = (float)gates_T[(512 + jj) * 32 + bl]  + bias_g[2];
                float go = (float)gates_T[(768 + jj) * 32 + bl]  + bias_g[3];
                const int ci = half * 8 + k;
                float cn = sigf(gf) * c_reg[ci] + sigf(gi) * tanhf(gg);
                float hn = sigf(go) * tanhf(cn);
                float outv = m ? hn : 0.0f;
                if (m) { c_reg[ci] = cn; h_lds[b * 256 + jj] = (__bf16)hn; }
                int pos = (dir == 0) ? t : (m ? (len - 1 - t) : t);
                int col = (dir == 0) ? jj : (256 + jj);
                lstm_out[((size_t)b * 512 + pos) * 512 + col] = (__bf16)outv;
            }
            __syncthreads();
        }
    }
}

// ---------------- feats = lstm_out @ w_h2t^T + b_h2t  (32768x32, K=512) ----------------
__global__ __launch_bounds__(256) void feats_kernel(const __bf16* __restrict__ lstm_out,
                                                    const __bf16* __restrict__ w2t,
                                                    const float* __restrict__ b2t,
                                                    float* __restrict__ feats) {
    const int tileId = blockIdx.x * 8 + (threadIdx.x >> 5);
    const int lane   = threadIdx.x & 31;
    const int tm = tileId >> 1;       // 0..2047
    const int nt = tileId & 1;        // 0..1
    const int kHalf = lane >> 4;
    const int n0 = lane & 15;
    const int mRow = tm * 16 + (lane & 15);
    const __bf16* arow = lstm_out + (size_t)mRow * 512 + kHalf * 16;
    v8f acc = {};
#pragma unroll 1
    for (int kk = 0; kk < 16; ++kk) {
        v16bf a = load_a_frag(arow + kk * 32);
        v16bf bfrag = *(const v16bf*)(w2t + (size_t)(nt * 16 + n0) * 512 + kk * 32 + kHalf * 16);
        acc = __builtin_amdgcn_wmma_f32_16x16x32_bf16(false, a, false, bfrag, (short)0, acc,
                                                      false, false);
    }
    const int mBase = tm * 16 + kHalf * 8;
    const int ncol  = nt * 16 + n0;
    const float bias = b2t[ncol];
#pragma unroll
    for (int r = 0; r < 8; ++r)
        feats[(size_t)(mBase + r) * 32 + ncol] = acc[r] + bias;
}

// ---------------- Viterbi: one wave32 per sequence, lane = tag ----------------
__global__ __launch_bounds__(1024) void viterbi_kernel(const float* __restrict__ feats,
                                                       const int* __restrict__ lengths,
                                                       const float* __restrict__ trans,
                                                       unsigned char* __restrict__ bp,
                                                       float* __restrict__ out) {
    const int wave = blockIdx.x * 32 + (threadIdx.x >> 5);
    const int lane = threadIdx.x & 31;
    if (wave >= 64) return;
    const int b = wave;
    float tr[32];
#pragma unroll
    for (int i = 0; i < 32; ++i) tr[i] = trans[lane * 32 + i];   // trans[new=lane][old=i]
    float fv = (lane == 30) ? 0.0f : -10000.0f;                  // START = 30
    const int len = lengths[b];
    unsigned char* bpb = bp + (size_t)b * 512 * 32;
#pragma unroll 1
    for (int t = 0; t < 512; ++t) {
        float feat = feats[((size_t)b * 512 + t) * 32 + lane];
        float best = -3.4e38f;
        int arg = 0;
#pragma unroll
        for (int i = 0; i < 32; ++i) {
            float s = __shfl(fv, i, 32) + tr[i];
            if (s > best) { best = s; arg = i; }
        }
        bool m = (t < len);
        fv = m ? (best + feat) : fv;
        bpb[t * 32 + lane] = (unsigned char)(m ? arg : lane);
    }
    float term = fv + trans[31 * 32 + lane];                     // STOP = 31
    float bestv = term;
    int besti = lane;
#pragma unroll
    for (int off = 16; off; off >>= 1) {
        float ov = __shfl_xor(bestv, off, 32);
        int   oi = __shfl_xor(besti, off, 32);
        if (ov > bestv || (ov == bestv && oi < besti)) { bestv = ov; besti = oi; }
    }
    if (lane == 0) {
        out[b] = bestv;
        float* path = out + 64 + (size_t)b * 512;
        int tag = besti;
        for (int t = 511; t >= 0; --t) {
            path[t] = (float)tag;
            tag = bpb[t * 32 + tag];
        }
    }
}

// ---------------- launch ----------------
extern "C" void kernel_launch(void* const* d_in, const int* in_sizes, int n_in,
                              void* d_out, int out_size, void* d_ws, size_t ws_size,
                              hipStream_t stream) {
    const int*   sent      = (const int*)d_in[0];
    const int*   lengths   = (const int*)d_in[1];
    const float* embedding = (const float*)d_in[2];
    const float* wih_f = (const float*)d_in[3];
    const float* whh_f = (const float*)d_in[4];
    const float* bih_f = (const float*)d_in[5];
    const float* bhh_f = (const float*)d_in[6];
    const float* wih_b = (const float*)d_in[7];
    const float* whh_b = (const float*)d_in[8];
    const float* bih_b = (const float*)d_in[9];
    const float* bhh_b = (const float*)d_in[10];
    const float* w_h2t = (const float*)d_in[11];
    const float* b_h2t = (const float*)d_in[12];
    const float* h0    = (const float*)d_in[13];
    const float* c0    = (const float*)d_in[14];
    const float* trans = (const float*)d_in[15];

    char* ws = (char*)d_ws;
    __bf16* embF  = (__bf16*)(ws + OFF_EMBF);
    __bf16* embB  = (__bf16*)(ws + OFF_EMBB);
    __bf16* wihF  = (__bf16*)(ws + OFF_WIHF);
    __bf16* whhF  = (__bf16*)(ws + OFF_WHHF);
    __bf16* wihB  = (__bf16*)(ws + OFF_WIHB);
    __bf16* whhB  = (__bf16*)(ws + OFF_WHHB);
    __bf16* w2t   = (__bf16*)(ws + OFF_W2T);
    __bf16* lstmo = (__bf16*)(ws + OFF_LSTM);
    float*  feats = (float*)(ws + OFF_FEATS);
    unsigned char* bp = (unsigned char*)(ws + OFF_BP);

    cvt_kernel<<<1024, 256, 0, stream>>>(wih_f, wihF, 262144);
    cvt_kernel<<<1024, 256, 0, stream>>>(whh_f, whhF, 262144);
    cvt_kernel<<<1024, 256, 0, stream>>>(wih_b, wihB, 262144);
    cvt_kernel<<<1024, 256, 0, stream>>>(whh_b, whhB, 262144);
    cvt_kernel<<<64, 256, 0, stream>>>(w_h2t, w2t, 16384);

    embed_kernel<<<4096, 256, 0, stream>>>(sent, lengths, embedding, embF, embB);

    lstm_kernel<<<2, 1024, 0, stream>>>(embF, embB, wihF, whhF, wihB, whhB,
                                        bih_f, bhh_f, bih_b, bhh_b,
                                        h0, c0, lengths, lstmo);

    feats_kernel<<<512, 256, 0, stream>>>(lstmo, w2t, b_h2t, feats);

    viterbi_kernel<<<2, 1024, 0, stream>>>(feats, lengths, trans, bp, (float*)d_out);
}
